// GCN_12515534700679
// MI455X (gfx1250) — matrix operand order repaired and checked by reading, not compiled, not run on hardware
//
#include <hip/hip_runtime.h>
#include <hip/hip_bf16.h>

typedef __attribute__((ext_vector_type(16))) __bf16 v16bf;
typedef __attribute__((ext_vector_type(8)))  float  v8f;

#define N_NODES 50000
#define F_IN    512
#define F_OUT   96
#define N_EDGES 800000

// Padded k-stride (bf16 elems) for the LDS column-major W tile:
// byte stride 1040 -> dword stride 260 -> bank step 4 -> the 16 lanes of each
// half-wave land on 16 distinct banks when fetching B fragments.
#define WT_STRIDE 520

// ---------------------------------------------------------------------------
// Zero-init of the output accumulator (d_out is poisoned by the harness).
__global__ void gcn_zero(float* __restrict__ out, int n) {
  int i = blockIdx.x * blockDim.x + threadIdx.x;
  if (i < n) out[i] = 0.0f;
}

// ---------------------------------------------------------------------------
// support = X @ W  via v_wmma_f32_16x16x32_bf16 (fp32 accumulate).
// Block = 256 threads = 8 waves; each wave computes a 16-row strip x 96 cols.
__global__ void gcn_gemm(const float* __restrict__ X,
                         const float* __restrict__ W,
                         float* __restrict__ support) {
  extern __shared__ __bf16 lds_wt[];  // [F_OUT][WT_STRIDE] column-major W (bf16)

  // Cooperative transpose + fp32->bf16 convert of W[k][n] -> lds_wt[n][k].
  for (int i = threadIdx.x; i < F_IN * F_OUT; i += blockDim.x) {
    int k = i / F_OUT;
    int n = i % F_OUT;
    lds_wt[n * WT_STRIDE + k] = (__bf16)W[i];
  }
  __syncthreads();

  const int wave   = threadIdx.x >> 5;
  const int lane   = threadIdx.x & 31;
  const int lanelo = lane & 15;
  const int half   = lane >> 4;

  const int row_base = blockIdx.x * 128 + wave * 16;
  int m  = row_base + lanelo;
  int mc = (m < N_NODES) ? m : (N_NODES - 1);     // clamp loads; stores guarded
  const float* __restrict__ xrow = X + (size_t)mc * F_IN;

  v8f acc[6] = {};

  for (int k0 = 0; k0 < F_IN; k0 += 32) {
    // Prefetch the K-block two steps ahead of this row (global_prefetch_b8).
    if (k0 + 64 <= F_IN - 32) __builtin_prefetch(xrow + k0 + 64, 0, 3);

    // ---- A fragment: 16x32 bf16, contiguous k order (same as B):
    // VGPR v (0..7): K = half*16 + 2v, +1
    v16bf a;
#pragma unroll
    for (int vg = 0; vg < 8; ++vg) {
      int K = half * 16 + 2 * vg;
      float2 p = *(const float2*)(xrow + k0 + K);
      a[2 * vg]     = (__bf16)p.x;
      a[2 * vg + 1] = (__bf16)p.y;
    }

    // ---- 6 N-tiles: B fragment 32x16 bf16 from LDS, contiguous k order (same as A):
    // N = lane%16 ; VGPR v: K = half*16 + 2v, +1
#pragma unroll
    for (int t = 0; t < 6; ++t) {
      const int n = t * 16 + lanelo;
      const __bf16* __restrict__ wc = lds_wt + n * WT_STRIDE + k0 + half * 16;
      v16bf b;
#pragma unroll
      for (int vg = 0; vg < 8; ++vg) {
        b[2 * vg]     = wc[2 * vg];
        b[2 * vg + 1] = wc[2 * vg + 1];
      }
      acc[t] = __builtin_amdgcn_wmma_f32_16x16x32_bf16(
          /*neg_a=*/false, a, /*neg_b=*/false, b,
          /*c_mod=*/(short)0, acc[t], /*reuse_a=*/false, /*reuse_b=*/false);
    }
  }

  // ---- Store D (f32 16x16 layout): VGPR r -> M = r + half*8, N = lane%16.
  // Wave-uniform fast path: full 16-row strip in range -> straight-line
  // global_store_b32 with immediate offsets off one base address.
  float* __restrict__ outp =
      support + (size_t)(row_base + half * 8) * F_OUT + lanelo;
  if (row_base + 16 <= N_NODES) {
#pragma unroll
    for (int t = 0; t < 6; ++t) {
#pragma unroll
      for (int r = 0; r < 8; ++r) {
        outp[(size_t)r * F_OUT + t * 16] = acc[t][r];
      }
    }
  } else {
    // Tail strip: per-row guard.
#pragma unroll
    for (int t = 0; t < 6; ++t) {
#pragma unroll
      for (int r = 0; r < 8; ++r) {
        if (row_base + half * 8 + r < N_NODES)
          outp[(size_t)r * F_OUT + t * 16] = acc[t][r];
      }
    }
  }
}

// ---------------------------------------------------------------------------
// out[row] += val * support[col]  (one wave per edge; gathers hit L2,
// scatter via global_atomic_add_f32).
__global__ void gcn_scatter(const int* __restrict__ erow,
                            const int* __restrict__ ecol,
                            const float* __restrict__ evalv,
                            const float* __restrict__ support,
                            float* __restrict__ out, int E) {
  int e = blockIdx.x * (blockDim.x >> 5) + (threadIdx.x >> 5);
  if (e >= E) return;
  int lane = threadIdx.x & 31;
  int r = erow[e];
  int c = ecol[e];
  float v = evalv[e];
  const float* __restrict__ src = support + (size_t)c * F_OUT;
  float* __restrict__ dst = out + (size_t)r * F_OUT;
#pragma unroll
  for (int f = 0; f < 3; ++f) {
    int j = lane + 32 * f;
    atomicAdd(dst + j, v * src[j]);
  }
}

// ---------------------------------------------------------------------------
__global__ void gcn_relu(float* __restrict__ out, int n) {
  int i = blockIdx.x * blockDim.x + threadIdx.x;
  if (i < n) out[i] = fmaxf(out[i], 0.0f);
}

// ---------------------------------------------------------------------------
extern "C" void kernel_launch(void* const* d_in, const int* in_sizes, int n_in,
                              void* d_out, int out_size, void* d_ws, size_t ws_size,
                              hipStream_t stream) {
  const float* X     = (const float*)d_in[0];
  const int*   erow  = (const int*)d_in[1];
  const int*   ecol  = (const int*)d_in[2];
  const float* evalv = (const float*)d_in[3];
  const float* W     = (const float*)d_in[4];
  float* out     = (float*)d_out;
  float* support = (float*)d_ws;   // 50000*96 fp32 = 19.2 MB scratch

  const int nOut = N_NODES * F_OUT;

  gcn_zero<<<(nOut + 255) / 256, 256, 0, stream>>>(out, nOut);

  const int gemm_blocks = (N_NODES + 127) / 128;
  const size_t lds_bytes = (size_t)F_OUT * WT_STRIDE * sizeof(__bf16); // ~97.5 KB
  gcn_gemm<<<gemm_blocks, 256, lds_bytes, stream>>>(X, W, support);

  const int waves_per_block = 8;
  gcn_scatter<<<(N_EDGES + waves_per_block - 1) / waves_per_block, 256, 0, stream>>>(
      erow, ecol, evalv, support, out, N_EDGES);

  gcn_relu<<<(nOut + 255) / 256, 256, 0, stream>>>(out, nOut);
}
